// OwnSasrecModel_50440095924479
// MI455X (gfx1250) — matrix operand order repaired and checked by reading, not compiled, not run on hardware
//
#include <hip/hip_runtime.h>
#include <hip/hip_bf16.h>

// ---------------------------------------------------------------------------
// SASRec forward for MI455X (gfx1250).
// All matmuls: v_wmma_f32_16x16x32_bf16 (bf16 operands, f32 accumulation).
// All WMMA operands staged as bf16 so fragment loads are contiguous b128s;
// all bounds handling via index clamping (no exec-masked loads).
// ---------------------------------------------------------------------------

typedef __attribute__((ext_vector_type(16))) __bf16 v16bf;
typedef __attribute__((ext_vector_type(8)))  __bf16 v8bf;
typedef __attribute__((ext_vector_type(8)))  float  v8f;

#define CAT16(lo, hi) __builtin_shufflevector((lo), (hi), \
    0, 1, 2, 3, 4, 5, 6, 7, 8, 9, 10, 11, 12, 13, 14, 15)

#define BB      512
#define SS      200
#define DD      256
#define HH      4
#define DHH     64
#define NITEMS  100000
#define NBLK    2
#define QTILES  13                 // ceil(S/16)
#define SCALE_F 0.125f             // 1/sqrt(64)
#define NEGF    (-4294967295.0f)   // reference NEG

static __device__ __forceinline__ int imin(int a, int b) { return a < b ? a : b; }

static __device__ __forceinline__ v8f wmma_bf16(v16bf a, v16bf b, v8f c) {
    return __builtin_amdgcn_wmma_f32_16x16x32_bf16(
        /*neg_a=*/false, a, /*neg_b=*/false, b,
        /*c_mod=*/(short)0, c, /*reuse_a=*/false, /*reuse_b=*/false);
}

// ---------------------------------------------------------------------------
// seq[b,s,:] = item_emb[ids[b,s]] + pos_emb[s]   (f32 + bf16 copies)
// ---------------------------------------------------------------------------
__global__ __launch_bounds__(256) void embed_kernel(
    const int* __restrict__ ids, const float* __restrict__ item_emb,
    const float* __restrict__ pos_emb, float* __restrict__ seq,
    __bf16* __restrict__ seq_bf, int total)
{
    int i = blockIdx.x * 256 + threadIdx.x;
    if (i >= total) return;
    int d  = i % DD;
    int bs = i / DD;
    int s  = bs % SS;
    float v = item_emb[(size_t)ids[bs] * DD + d] + pos_emb[(size_t)s * DD + d];
    seq[i]    = v;
    seq_bf[i] = (__bf16)v;
}

// ---------------------------------------------------------------------------
// Weight prep: transpose [K,N] f32 -> [N,K] bf16 (so GEMM B-frags are
// contiguous along K), and plain f32 -> bf16 convert for out_item_emb.
// ---------------------------------------------------------------------------
__global__ __launch_bounds__(256) void wtrans_kernel(
    const float* __restrict__ W, __bf16* __restrict__ out, int K, int N)
{
    int i = blockIdx.x * 256 + threadIdx.x;   // index into [N,K] output
    if (i >= K * N) return;
    int n = i / K, k = i % K;
    out[i] = (__bf16)W[(size_t)k * N + n];
}

__global__ __launch_bounds__(256) void conv_kernel(
    const float* __restrict__ in, __bf16* __restrict__ out, int total)
{
    int i = blockIdx.x * 256 + threadIdx.x;
    if (i >= total) return;
    out[i] = (__bf16)in[i];
}

// ---------------------------------------------------------------------------
// LayerNorm (eps=1e-3), one wave per row of 256. Optional fused add
// (x2 = a + b, raw sum to sum_out), strided a-rows for the final
// seq[:, -1, :] LN, and optional bf16 copy of the normalized output.
// NOTE: norm_out may alias b (read-all-then-write per row) -> no restrict.
// ---------------------------------------------------------------------------
__global__ __launch_bounds__(256) void ln_kernel(
    const float* a, long long astride, const float* b,
    const float* __restrict__ g, const float* __restrict__ be,
    float* sum_out, float* norm_out, __bf16* norm_bf, int rows)
{
    int lane = threadIdx.x & 31;
    int row  = blockIdx.x * 8 + (threadIdx.x >> 5);
    if (row >= rows) return;
    const float* ar = a + (size_t)row * (size_t)astride + lane * 8;
    v8f av = *(const v8f*)ar;
    if (b) {
        v8f bv = *(const v8f*)(b + (size_t)row * DD + lane * 8);
#pragma unroll
        for (int j = 0; j < 8; j++) av[j] += bv[j];
    }
    float s = 0.f;
#pragma unroll
    for (int j = 0; j < 8; j++) s += av[j];
#pragma unroll
    for (int off = 16; off > 0; off >>= 1) s += __shfl_xor(s, off, 32);
    float mean = s * (1.0f / DD);
    float vs = 0.f;
#pragma unroll
    for (int j = 0; j < 8; j++) { float d = av[j] - mean; vs += d * d; }
#pragma unroll
    for (int off = 16; off > 0; off >>= 1) vs += __shfl_xor(vs, off, 32);
    float rstd = rsqrtf(vs * (1.0f / DD) + 1e-3f);

    v8f gv = *(const v8f*)(g + lane * 8);
    v8f bev = *(const v8f*)(be + lane * 8);
    v8f ov;
#pragma unroll
    for (int j = 0; j < 8; j++) ov[j] = (av[j] - mean) * rstd * gv[j] + bev[j];
    if (sum_out) *(v8f*)(sum_out + (size_t)row * DD + lane * 8) = av;
    *(v8f*)(norm_out + (size_t)row * DD + lane * 8) = ov;
    if (norm_bf) {
        v8bf p;
#pragma unroll
        for (int j = 0; j < 8; j++) p[j] = (__bf16)ov[j];
        *(v8bf*)(norm_bf + (size_t)row * DD + lane * 8) = p;
    }
}

// ---------------------------------------------------------------------------
// GEMM: C[M,N] = A[M,K] * W^T + bias.  A bf16 [M,K] row-major, W bf16 [N,K]
// row-major (pre-transposed weights / out_item_emb), so both fragment types
// are contiguous b128 loads. Each wave: 32x32 tile = 2x2 WMMA frags.
// Ragged N handled by clamped loads + store guards (no exec-masked loads).
// Epilogue outputs: f32 [M,N], bf16 [M,N], and/or bf16 V^T [B,H,DH,S].
// Fused: bias, relu, residual (post-relu), pad-row masking.
// ---------------------------------------------------------------------------
__global__ __launch_bounds__(256) void gemm_kernel(
    const __bf16* __restrict__ A, const __bf16* __restrict__ W,
    const float* __restrict__ bias, const float* __restrict__ Rres,
    const int* __restrict__ ids,
    float* __restrict__ outF, __bf16* __restrict__ outB, __bf16* __restrict__ outVT,
    int M, int N, int Kd, int do_relu)
{
    int lane = threadIdx.x & 31;
    int hf = lane >> 4;
    int ln = lane & 15;
    int tn = (N + 31) >> 5;
    int tm = M >> 5;
    int w  = blockIdx.x * 8 + (threadIdx.x >> 5);
    if (w >= tm * tn) return;                 // wave-uniform
    int m0 = (w / tn) * 32;
    int n0 = (w % tn) * 32;

    const __bf16* arow[2];
    const __bf16* wrow[2];
#pragma unroll
    for (int i = 0; i < 2; i++)
        arow[i] = A + (size_t)(m0 + i * 16 + ln) * Kd + (hf ? 8 : 0);
#pragma unroll
    for (int j = 0; j < 2; j++) {
        int colc = imin(n0 + j * 16 + ln, N - 1);     // clamped: load real data
        wrow[j] = W + (size_t)colc * Kd + (hf ? 8 : 0);
    }

    v8f acc[2][2];
#pragma unroll
    for (int i = 0; i < 2; i++)
#pragma unroll
        for (int j = 0; j < 2; j++)
#pragma unroll
            for (int r = 0; r < 8; r++) acc[i][j][r] = 0.f;

    for (int kb = 0; kb < Kd; kb += 32) {
        if (kb + 32 < Kd) {                            // global_prefetch_b8
            __builtin_prefetch(arow[0] + kb + 32, 0, 3);
            __builtin_prefetch(wrow[0] + kb + 32, 0, 3);
        }
        v16bf af[2], wf[2];
#pragma unroll
        for (int i = 0; i < 2; i++)
            af[i] = CAT16(*(const v8bf*)(arow[i] + kb),
                          *(const v8bf*)(arow[i] + kb + 16));
#pragma unroll
        for (int j = 0; j < 2; j++)
            wf[j] = CAT16(*(const v8bf*)(wrow[j] + kb),
                          *(const v8bf*)(wrow[j] + kb + 16));
#pragma unroll
        for (int i = 0; i < 2; i++)
#pragma unroll
            for (int j = 0; j < 2; j++)
                acc[i][j] = wmma_bf16(af[i], wf[j], acc[i][j]);
    }

    // Epilogue. C layout: lane holds column ln, vgpr r holds row (hf?8:0)+r.
#pragma unroll
    for (int i = 0; i < 2; i++) {
#pragma unroll
        for (int j = 0; j < 2; j++) {
            int col  = n0 + j * 16 + ln;
            int colc = imin(col, N - 1);
            bool vc  = col < N;
            float bv = bias ? bias[colc] : 0.f;
            float vr[8];
#pragma unroll
            for (int r = 0; r < 8; r++) {
                int row = m0 + i * 16 + (hf ? 8 : 0) + r;
                float v = acc[i][j][r] + bv;
                if (do_relu) v = fmaxf(v, 0.f);
                if (Rres) v += Rres[(size_t)row * N + colc];
                if (ids)  v *= (ids[row] != NITEMS) ? 1.f : 0.f;
                vr[r] = v;
                if (outF && vc) outF[(size_t)row * N + col] = v;
                if (outB && vc) outB[(size_t)row * N + col] = (__bf16)v;
            }
            if (outVT) {
                // V^T store: [B,H,DH,S] bf16; 8-row run never crosses a batch
                // boundary (runs 8-aligned, S multiple of 8) -> one b128 store.
                int row0 = m0 + i * 16 + (hf ? 8 : 0);
                int bb = row0 / SS, s0 = row0 % SS;
                int hh = col / DHH, dd = col % DHH;
                v8bf p;
#pragma unroll
                for (int r = 0; r < 8; r++) p[r] = (__bf16)vr[r];
                *(v8bf*)(outVT + ((size_t)(bb * HH + hh) * DHH + dd) * SS + s0) = p;
            }
        }
    }
}

// ---------------------------------------------------------------------------
// Causal flash attention, computed TRANSPOSED so online-softmax stats are
// per-lane-column scalars and the P relayout is one shfl_xor(16):
//   S^T = K * Q^T ,  o^T = V^T * P^T
// Q,K: bf16 [B*S, D] row-major. V: bf16 [B,H,DH,S] (pre-transposed by GEMM).
// One wave per (b, h, 16-query tile). All loads index-clamped (no guards).
// ---------------------------------------------------------------------------
__global__ __launch_bounds__(32) void attn_kernel(
    const __bf16* __restrict__ Qb, const __bf16* __restrict__ Kb,
    const __bf16* __restrict__ Vt, float* __restrict__ O)
{
    int idx = blockIdx.x;
    int qt = idx % QTILES; idx /= QTILES;
    int h  = idx % HH;
    int b  = idx / HH;
    int lane = threadIdx.x & 31;
    int hf = lane >> 4;
    int ln = lane & 15;
    size_t base = (size_t)b * SS * DD + (size_t)h * DHH;
    int q0 = qt * 16;
    int qrow = q0 + ln;
    int qrc  = imin(qrow, SS - 1);

    // Q^T B-fragments (dh = 64 -> two 32-wide k-steps)
    v16bf qf[2];
#pragma unroll
    for (int ks = 0; ks < 2; ks++) {
        const __bf16* qr = Qb + base + (size_t)qrc * DD + ks * 32 + (hf ? 8 : 0);
        qf[ks] = CAT16(*(const v8bf*)qr, *(const v8bf*)(qr + 16));
    }
    // V^T row pointers: lane's d-row per d-tile
    const __bf16* vtp[4];
#pragma unroll
    for (int dt = 0; dt < 4; dt++)
        vtp[dt] = Vt + ((size_t)(b * HH + h) * DHH + dt * 16 + ln) * SS + (hf ? 16 : 0);

    float m_i = -3.0e38f, l_i = 0.f;
    v8f oacc[4];
#pragma unroll
    for (int dt = 0; dt < 4; dt++)
#pragma unroll
        for (int r = 0; r < 8; r++) oacc[dt][r] = 0.f;

    int nch = (q0 + 15) / 32 + 1;              // causal: keys <= q0+15 only
    for (int c = 0; c < nch; c++) {
        int k0 = c * 32;
        v8f st[2];
#pragma unroll
        for (int t = 0; t < 2; t++) {
#pragma unroll
            for (int r = 0; r < 8; r++) st[t][r] = 0.f;
            int krc = imin(k0 + t * 16 + ln, SS - 1);
#pragma unroll
            for (int ks = 0; ks < 2; ks++) {
                const __bf16* kr = Kb + base + (size_t)krc * DD + ks * 32 + (hf ? 8 : 0);
                v16bf kf = CAT16(*(const v8bf*)kr, *(const v8bf*)(kr + 16));
                st[t] = wmma_bf16(kf, qf[ks], st[t]);
            }
        }
        // scale + causal/pad mask (register-only selects)
        float mx = m_i;
#pragma unroll
        for (int t = 0; t < 2; t++)
#pragma unroll
            for (int r = 0; r < 8; r++) {
                int ki = k0 + t * 16 + (hf ? 8 : 0) + r;
                float sc = st[t][r] * SCALE_F;
                sc = (ki <= qrow && ki < SS) ? sc : NEGF;
                st[t][r] = sc;
                mx = fmaxf(mx, sc);
            }
        mx = fmaxf(mx, __shfl_xor(mx, 16, 32));
        float corr = __expf(m_i - mx);
        m_i = mx;
        float rs = 0.f;
#pragma unroll
        for (int t = 0; t < 2; t++)
#pragma unroll
            for (int r = 0; r < 8; r++) {
                float p = __expf(st[t][r] - mx);
                st[t][r] = p;
                rs += p;
            }
        rs += __shfl_xor(rs, 16, 32);
        l_i = l_i * corr + rs;

        // P^T: C-layout -> B-fragment via single half-swap
        float sw0[8], sw1[8];
#pragma unroll
        for (int r = 0; r < 8; r++) {
            sw0[r] = __shfl_xor(st[0][r], 16, 32);
            sw1[r] = __shfl_xor(st[1][r], 16, 32);
        }
        v16bf pf;
#pragma unroll
        for (int e = 0; e < 8; e++) {
            pf[e]     = (__bf16)(hf ? sw1[e]   : st[0][e]);  // keys k0+16*hf+0..7
            pf[8 + e] = (__bf16)(hf ? st[1][e] : sw0[e]);    // keys k0+16*hf+8..15
        }
        // o^T += V^T * P^T (contiguous b128 loads from pre-transposed V)
#pragma unroll
        for (int dt = 0; dt < 4; dt++) {
#pragma unroll
            for (int r = 0; r < 8; r++) oacc[dt][r] *= corr;
            const __bf16* vr = vtp[dt] + k0;
            v16bf vf = CAT16(*(const v8bf*)vr, *(const v8bf*)(vr + 8));
            oacc[dt] = wmma_bf16(vf, pf, oacc[dt]);
        }
    }

    if (qrow < SS) {
        float inv = 1.0f / l_i;
        float* orow = O + base + (size_t)qrow * DD;
#pragma unroll
        for (int dt = 0; dt < 4; dt++) {
            v8f res;
#pragma unroll
            for (int r = 0; r < 8; r++) res[r] = oacc[dt][r] * inv;
            *(v8f*)(orow + dt * 16 + (hf ? 8 : 0)) = res;
        }
    }
}

// ---------------------------------------------------------------------------
extern "C" void kernel_launch(void* const* d_in, const int* in_sizes, int n_in,
                              void* d_out, int out_size, void* d_ws, size_t ws_size,
                              hipStream_t stream) {
    (void)in_sizes; (void)n_in; (void)out_size; (void)ws_size;
    const int*   ids      = (const int*)  d_in[0];
    const float* item_emb = (const float*)d_in[1];
    const float* pos_emb  = (const float*)d_in[2];
    const float* ln1_g    = (const float*)d_in[3];
    const float* ln1_b    = (const float*)d_in[4];
    const float* wq       = (const float*)d_in[5];
    const float* bq       = (const float*)d_in[6];
    const float* wk       = (const float*)d_in[7];
    const float* bk       = (const float*)d_in[8];
    const float* wv       = (const float*)d_in[9];
    const float* bv       = (const float*)d_in[10];
    const float* ln2_g    = (const float*)d_in[11];
    const float* ln2_b    = (const float*)d_in[12];
    const float* wd       = (const float*)d_in[13];
    const float* bd       = (const float*)d_in[14];
    const float* seq_g    = (const float*)d_in[15];
    const float* seq_b    = (const float*)d_in[16];
    const float* oie      = (const float*)d_in[17];
    float* out = (float*)d_out;

    const size_t NBSD = (size_t)BB * SS * DD;        // 26,214,400
    // f32 region (~420 MB)
    float* seq  = (float*)d_ws;
    float* xn   = seq  + NBSD;
    float* x2   = xn   + NBSD;
    float* obuf = x2   + NBSD;
    float* semb = obuf + NBSD;                       // [B, D]
    // bf16 region (~315 MB), 16B-aligned by construction
    __bf16* seq_bf = (__bf16*)(semb + (size_t)BB * DD);
    __bf16* xn_bf  = seq_bf + NBSD;
    __bf16* q_bf   = xn_bf  + NBSD;
    __bf16* k_bf   = q_bf   + NBSD;
    __bf16* vt_bf  = k_bf   + NBSD;                  // + 64 elem tail pad
    __bf16* oie_bf = vt_bf  + NBSD + 64;
    __bf16* wt_bf  = oie_bf + (size_t)NITEMS * DD;   // 8 x [N=256,K=256]
    __bf16* semb_bf = wt_bf + (size_t)8 * DD * DD;

    const int BS = BB * SS;
    {
        int total = BS * DD;
        embed_kernel<<<(total + 255) / 256, 256, 0, stream>>>(
            ids, item_emb, pos_emb, seq, seq_bf, total);
    }
    // weight prep: transpose+convert the 8 projection weights, convert oie
    const float* wsrc[8] = { wq, wk, wv, wd, wq + DD * DD, wk + DD * DD,
                             wv + DD * DD, wd + DD * DD };
    for (int t = 0; t < 8; t++)
        wtrans_kernel<<<(DD * DD + 255) / 256, 256, 0, stream>>>(
            wsrc[t], wt_bf + (size_t)t * DD * DD, DD, DD);
    conv_kernel<<<(NITEMS * DD + 255) / 256, 256, 0, stream>>>(
        oie, oie_bf, NITEMS * DD);

    const int lnBlocks   = (BS + 7) / 8;
    const int gemmBlocks = ((BS / 32) * (DD / 32) + 7) / 8;

    for (int i = 0; i < NBLK; i++) {
        const __bf16* wtq = wt_bf + (size_t)(i * 4 + 0) * DD * DD;
        const __bf16* wtk = wt_bf + (size_t)(i * 4 + 1) * DD * DD;
        const __bf16* wtv = wt_bf + (size_t)(i * 4 + 2) * DD * DD;
        const __bf16* wtd = wt_bf + (size_t)(i * 4 + 3) * DD * DD;
        // x = ln1(seq)
        ln_kernel<<<lnBlocks, 256, 0, stream>>>(
            seq, DD, nullptr, ln1_g + i * DD, ln1_b + i * DD,
            nullptr, xn, xn_bf, BS);
        // Q from normed x; K,V from raw seq (V stored transposed per-head)
        gemm_kernel<<<gemmBlocks, 256, 0, stream>>>(
            xn_bf,  wtq, bq + i * DD, nullptr, nullptr,
            nullptr, q_bf, nullptr, BS, DD, DD, 0);
        gemm_kernel<<<gemmBlocks, 256, 0, stream>>>(
            seq_bf, wtk, bk + i * DD, nullptr, nullptr,
            nullptr, k_bf, nullptr, BS, DD, DD, 0);
        gemm_kernel<<<gemmBlocks, 256, 0, stream>>>(
            seq_bf, wtv, bv + i * DD, nullptr, nullptr,
            nullptr, nullptr, vt_bf, BS, DD, DD, 0);
        // o = softmax(mask(q k^T / sqrt(dh))) v
        attn_kernel<<<BB * HH * QTILES, 32, 0, stream>>>(q_bf, k_bf, vt_bf, obuf);
        // x2 = o + x ; xn = ln2(x2) (+ bf16 copy)
        ln_kernel<<<lnBlocks, 256, 0, stream>>>(
            obuf, DD, xn, ln2_g + i * DD, ln2_b + i * DD, x2, xn, xn_bf, BS);
        // seq = (relu(xn @ wd + bd) + x2) * mask  (f32 + bf16 copies)
        gemm_kernel<<<gemmBlocks, 256, 0, stream>>>(
            xn_bf, wtd, bd + i * DD, x2, ids, seq, seq_bf, nullptr, BS, DD, DD, 1);
    }
    // seq_emb = ln(seq[:, -1, :])  (strided rows)
    ln_kernel<<<(BB + 7) / 8, 256, 0, stream>>>(
        seq + (size_t)(SS - 1) * DD, (long long)SS * DD, nullptr,
        seq_g, seq_b, nullptr, semb, semb_bf, BB);
    // logits = seq_emb @ out_item_emb^T  (oie already [N,K] row-major)
    int ftiles = (BB / 32) * ((NITEMS + 31) / 32);
    gemm_kernel<<<(ftiles + 7) / 8, 256, 0, stream>>>(
        semb_bf, oie_bf, nullptr, nullptr, nullptr,
        out, nullptr, nullptr, BB, NITEMS, DD, 0);
}
